// Model_59304908423840
// MI455X (gfx1250) — compile-verified, matrix operand order checked by reading, not run on hardware
//
#include <hip/hip_runtime.h>
#include <hip/hip_bf16.h>
#include <math.h>

// ---------------- problem constants ----------------
#define BATCH 64
#define SLEN  512
#define HDIM  768
#define TNUM  9
#define TP    16                 // padded tag dim
#define ROWS  (BATCH * SLEN)     // 32768
#define CHUNKS 32                // scan chunks over S
#define CLEN   16                // steps per chunk (CHUNKS*CLEN == SLEN)
#define NEG_BIG (-1.0e30f)       // finite "-inf" so lse of all-empty stays NaN-free

typedef __attribute__((ext_vector_type(16))) __bf16 v16bf;
typedef __attribute__((ext_vector_type(8)))  float  v8f;

// =====================================================================
// Kernel 1: emissions[r][t] = sum_k x[r][k] * w[t][k] + b[t]
// One wave per 16-row tile.  A = x tile (16x32), B = w^T tile (32x16).
// fp32 -> (bf16 hi + bf16 lo) split, 3 WMMAs/step for ~fp32 accuracy.
// HBM-bound: 100 MB of encoder_out @ 23.3 TB/s ~= 4.3 us floor.
// =====================================================================
__global__ __launch_bounds__(128) void emissions_wmma(
    const float* __restrict__ x,    // [ROWS][HDIM]
    const float* __restrict__ w,    // [TNUM][HDIM]
    const float* __restrict__ bias, // [TNUM]
    float* __restrict__ em)         // [ROWS][TP]
{
    const int lane = threadIdx.x & 31;
    const int wave = threadIdx.x >> 5;
    const int tile = blockIdx.x * 4 + wave;
    const int rowBase = tile * 16;
    const int m    = lane & 15;   // A row / B col for this lane
    const int half = lane >> 4;   // lane group 0 or 1

    const float* arow = x + (size_t)(rowBase + m) * HDIM;
    const float* brow = w + (size_t)((m < TNUM) ? m : 0) * HDIM;
    const bool bvalid = (m < TNUM);

    v8f acc = {};

    for (int kk = 0; kk < HDIM; kk += 32) {
        __builtin_prefetch(arow + kk + 64, 0, 0);

        const int k0 = kk + 8 * half;
        const int k1 = kk + 16 + 8 * half;

        float av[16], bv[16];
        {
            float4 a0 = *(const float4*)(arow + k0);
            float4 a1 = *(const float4*)(arow + k0 + 4);
            float4 a2 = *(const float4*)(arow + k1);
            float4 a3 = *(const float4*)(arow + k1 + 4);
            av[0]=a0.x; av[1]=a0.y; av[2]=a0.z;  av[3]=a0.w;
            av[4]=a1.x; av[5]=a1.y; av[6]=a1.z;  av[7]=a1.w;
            av[8]=a2.x; av[9]=a2.y; av[10]=a2.z; av[11]=a2.w;
            av[12]=a3.x; av[13]=a3.y; av[14]=a3.z; av[15]=a3.w;
        }
        if (bvalid) {
            float4 b0 = *(const float4*)(brow + k0);
            float4 b1 = *(const float4*)(brow + k0 + 4);
            float4 b2 = *(const float4*)(brow + k1);
            float4 b3 = *(const float4*)(brow + k1 + 4);
            bv[0]=b0.x; bv[1]=b0.y; bv[2]=b0.z;  bv[3]=b0.w;
            bv[4]=b1.x; bv[5]=b1.y; bv[6]=b1.z;  bv[7]=b1.w;
            bv[8]=b2.x; bv[9]=b2.y; bv[10]=b2.z; bv[11]=b2.w;
            bv[12]=b3.x; bv[13]=b3.y; bv[14]=b3.z; bv[15]=b3.w;
        } else {
            #pragma unroll
            for (int e = 0; e < 16; ++e) bv[e] = 0.0f;
        }

        v16bf ah, al, bh, bl;
        #pragma unroll
        for (int e = 0; e < 16; ++e) {
            float xa = av[e];
            __bf16 ha = (__bf16)xa;
            ah[e] = ha;
            al[e] = (__bf16)(xa - (float)ha);
            float xb = bv[e];
            __bf16 hb = (__bf16)xb;
            bh[e] = hb;
            bl[e] = (__bf16)(xb - (float)hb);
        }

        acc = __builtin_amdgcn_wmma_f32_16x16x32_bf16(false, ah, false, bh,
                                                      (short)0, acc, false, false);
        acc = __builtin_amdgcn_wmma_f32_16x16x32_bf16(false, ah, false, bl,
                                                      (short)0, acc, false, false);
        acc = __builtin_amdgcn_wmma_f32_16x16x32_bf16(false, al, false, bh,
                                                      (short)0, acc, false, false);
    }

    const int n = lane & 15;
    const float bb = (n < TNUM) ? bias[n] : 0.0f;
    #pragma unroll
    for (int j = 0; j < 8; ++j) {
        int r = rowBase + j + 8 * half;
        em[(size_t)r * TP + n] = acc[j] + bb;
    }
}

// =====================================================================
// Kernel 2: per-chunk transfer matrices in both semirings.
//   M_c = prod_s A_s   with A_s[p][t] = trans[p][t] + em_s[t]
//   (logsumexp,+) for the partition function, (max,+) for Viterbi.
//   Masked step == semiring identity (skip update).
// grid (CHUNKS, BATCH), 160 threads: (i = tid>>4, t = tid&15), 81 active.
// =====================================================================
__global__ __launch_bounds__(160) void chunk_mats(
    const float* __restrict__ em,     // [ROWS][TP]
    const int*   __restrict__ mask,   // [BATCH][SLEN]
    const float* __restrict__ trans,  // [TNUM][TNUM]
    float* __restrict__ MW,           // [BATCH][CHUNKS][TNUM][TP]
    float* __restrict__ VW)           // same shape
{
    const int c = blockIdx.x, b = blockIdx.y;
    const int tid = threadIdx.x;
    const int t = tid & 15;
    const int i = tid >> 4;                 // 0..9
    const bool act = (i < TNUM) && (t < TNUM);

    __shared__ float sh_tr[TNUM][TNUM];
    __shared__ float Ml[TNUM][TP];
    __shared__ float Vl[TNUM][TP];

    for (int idx = tid; idx < TNUM * TNUM; idx += 160)
        sh_tr[idx / TNUM][idx % TNUM] = trans[idx];
    if (i < TNUM) {
        float ini = (i == t) ? 0.0f : NEG_BIG;   // semiring identity matrix
        Ml[i][t] = ini;
        Vl[i][t] = ini;
    }
    __syncthreads();

    const int sFirst = (c == 0) ? 1 : CLEN * c;
    const int sLast  = CLEN * c + CLEN - 1;

    for (int s = sFirst; s <= sLast; ++s) {
        float nM = 0.0f, nV = 0.0f;
        bool upd = false;
        if (act) {
            const int mk = mask[b * SLEN + s];
            const float emt = em[((size_t)b * SLEN + s) * TP + t];
            float u[TNUM];
            float mx = NEG_BIG, bm = NEG_BIG;
            #pragma unroll
            for (int p = 0; p < TNUM; ++p) {
                float trpt = sh_tr[p][t];
                u[p] = Ml[i][p] + trpt;
                mx = fmaxf(mx, u[p]);
                bm = fmaxf(bm, Vl[i][p] + trpt);
            }
            float sm = 0.0f;
            #pragma unroll
            for (int p = 0; p < TNUM; ++p) sm += __expf(u[p] - mx);
            nM = mx + __logf(sm) + emt;
            nV = bm + emt;
            upd = (mk > 0);
        }
        __syncthreads();
        if (upd) { Ml[i][t] = nM; Vl[i][t] = nV; }
        __syncthreads();
    }

    if (i < TNUM) {
        size_t base = ((size_t)(b * CHUNKS + c) * TNUM + i) * TP + t;
        MW[base] = Ml[i][t];
        VW[base] = Vl[i][t];
    }
}

// =====================================================================
// Kernel 3: per-batch boundary pass (32 chunk applications) + numerator.
// wave0: alpha/valpha vector sweep over chunk matrices, denominator,
//        viterbi last-tag, and per-chunk viterbi entry vectors.
// wave1: gold-path numerator via 32-lane segmented scan over mask.
// =====================================================================
__global__ __launch_bounds__(64) void boundary_scan(
    const float* __restrict__ em,
    const int*   __restrict__ labels,
    const int*   __restrict__ mask,
    const float* __restrict__ start,
    const float* __restrict__ endt,
    const float* __restrict__ trans,
    const float* __restrict__ MW,
    const float* __restrict__ VW,
    float* __restrict__ ventry,       // [BATCH][CHUNKS][TP]
    float* __restrict__ ll,           // [BATCH]
    int*   __restrict__ lastTagW)     // [BATCH]
{
    const int b = blockIdx.x;
    const int tid = threadIdx.x;

    __shared__ float sh_den;
    __shared__ int   sh_lastT[32];
    __shared__ float sh_sum[32];

    const float* emb = em + (size_t)b * SLEN * TP;
    const int*   lb  = labels + (size_t)b * SLEN;
    const int*   mkp = mask   + (size_t)b * SLEN;

    if (tid < 32) {
        const int t = tid;
        const int tc = t & 15;
        float alpha  = (t < TNUM) ? start[t] + emb[t] : NEG_BIG;
        float valpha = alpha;

        for (int c = 0; c < CHUNKS; ++c) {
            if (t < TP) ventry[((size_t)b * CHUNKS + c) * TP + t] = valpha;
            const float* Mc = MW + ((size_t)(b * CHUNKS + c) * TNUM) * TP;
            const float* Vc = VW + ((size_t)(b * CHUNKS + c) * TNUM) * TP;

            float u[TNUM];
            float mx = NEG_BIG, vn = NEG_BIG;
            #pragma unroll
            for (int i = 0; i < TNUM; ++i) {
                float ai = __shfl(alpha,  i);
                float vi = __shfl(valpha, i);
                u[i] = ai + Mc[i * TP + tc];
                mx = fmaxf(mx, u[i]);
                vn = fmaxf(vn, vi + Vc[i * TP + tc]);
            }
            float sm = 0.0f;
            #pragma unroll
            for (int i = 0; i < TNUM; ++i) sm += __expf(u[i] - mx);
            float an = mx + __logf(sm);
            if (t < TNUM) { alpha = an; valpha = vn; }
        }

        float ft = (t < TNUM) ? alpha  + endt[t] : NEG_BIG;
        float fv = (t < TNUM) ? valpha + endt[t] : NEG_BIG;
        float fa[TNUM], fvv[TNUM];
        #pragma unroll
        for (int i = 0; i < TNUM; ++i) {
            fa[i]  = __shfl(ft, i);
            fvv[i] = __shfl(fv, i);
        }
        if (t == 0) {
            float mx = NEG_BIG;
            for (int i = 0; i < TNUM; ++i) mx = fmaxf(mx, fa[i]);
            float sm = 0.0f;
            for (int i = 0; i < TNUM; ++i) sm += __expf(fa[i] - mx);
            sh_den = mx + __logf(sm);
            float bm = NEG_BIG; int ba = 0;
            for (int i = 0; i < TNUM; ++i)
                if (fvv[i] > bm) { bm = fvv[i]; ba = i; }
            lastTagW[b] = ba;
        }
    } else {
        // numerator pass 1: last gold tag with mask>0 in my 16-step segment
        const int j = tid - 32;
        int lastT = -1;
        const int s0 = j * 16;
        for (int k = 0; k < 16; ++k) {
            int s = s0 + k;
            if (s >= 1 && mkp[s] > 0) lastT = lb[s];
        }
        sh_lastT[j] = lastT;
    }
    __syncthreads();

    if (tid >= 32) {
        const int j = tid - 32;
        const int tg0 = lb[0];
        int prev = tg0;
        for (int q = j - 1; q >= 0; --q)
            if (sh_lastT[q] >= 0) { prev = sh_lastT[q]; break; }
        float sum = 0.0f;
        const int s0 = j * 16;
        for (int k = 0; k < 16; ++k) {
            int s = s0 + k;
            if (s < 1) continue;
            int tg = lb[s];
            if (mkp[s] > 0) {
                sum += trans[prev * TNUM + tg] + emb[(size_t)s * TP + tg];
                prev = tg;
            }
        }
        sh_sum[j] = sum;
    }
    __syncthreads();

    if (tid == 0) {
        const int tg0 = lb[0];
        float num = start[tg0] + emb[tg0];
        for (int j = 0; j < 32; ++j) num += sh_sum[j];
        ll[b] = num - sh_den;
    }
}

// =====================================================================
// Kernel 4: per-chunk Viterbi recompute from exact boundary vectors:
// emit per-step backpointers + chunk-composed backtrace map H_c.
// grid (CHUNKS, BATCH), 1 wave.
// =====================================================================
__global__ __launch_bounds__(32) void bp_gen(
    const float* __restrict__ em,
    const int*   __restrict__ mask,
    const float* __restrict__ trans,
    const float* __restrict__ ventry,
    unsigned char* __restrict__ bpW,   // [ROWS][TP]
    unsigned char* __restrict__ HcW)   // [BATCH][CHUNKS][TP]
{
    const int c = blockIdx.x, b = blockIdx.y;
    const int t = threadIdx.x;
    const int tc = t & 15;
    const int tt = (t < TNUM) ? t : 0;

    __shared__ float sh_tr[TNUM][TNUM];
    __shared__ unsigned char bpl[CLEN][TP];

    for (int idx = t; idx < TNUM * TNUM; idx += 32)
        sh_tr[idx / TNUM][idx % TNUM] = trans[idx];
    __syncthreads();

    float v = ventry[((size_t)b * CHUNKS + c) * TP + tc];
    if (t >= TNUM) v = NEG_BIG;

    const int sFirst = (c == 0) ? 1 : CLEN * c;
    const int sLast  = CLEN * c + CLEN - 1;

    for (int s = sFirst; s <= sLast; ++s) {
        float bm = NEG_BIG; int ba = 0;
        #pragma unroll
        for (int p = 0; p < TNUM; ++p) {
            float vp = __shfl(v, p) + sh_tr[p][tt];
            if (vp > bm) { bm = vp; ba = p; }   // first max wins
        }
        const float emt = em[((size_t)b * SLEN + s) * TP + tc];
        const int mk = mask[b * SLEN + s];
        int nb = t;
        if (t < TNUM && mk > 0) { v = bm + emt; nb = ba; }
        if (t < TP) {
            unsigned char bb = (unsigned char)nb;
            bpl[s - CLEN * c][t] = bb;
            bpW[((size_t)b * SLEN + s) * TP + t] = bb;
        }
    }
    __syncthreads();

    if (t < TP) {
        int tmp = t;
        for (int s = sLast; s >= sFirst; --s) tmp = bpl[s - CLEN * c][tmp];
        HcW[((size_t)b * CHUNKS + c) * TP + t] = (unsigned char)tmp;
    }
}

// =====================================================================
// Kernel 5: chain the 32 chunk maps (in LDS) to get the path tag at
// every chunk boundary.  grid BATCH, 1 wave.
// =====================================================================
__global__ __launch_bounds__(32) void chain_kernel(
    const unsigned char* __restrict__ HcW,
    const int* __restrict__ lastTagW,
    unsigned char* __restrict__ tagEndW)   // [BATCH][CHUNKS]
{
    const int b = blockIdx.x;
    const int t = threadIdx.x;
    __shared__ unsigned char H[CHUNKS][TP];
    for (int idx = t; idx < CHUNKS * TP; idx += 32)
        H[idx >> 4][idx & 15] = HcW[(size_t)b * CHUNKS * TP + idx];
    __syncthreads();
    if (t == 0) {
        int tag = lastTagW[b];
        for (int c = CHUNKS - 1; c >= 0; --c) {
            tagEndW[(size_t)b * CHUNKS + c] = (unsigned char)tag;
            tag = H[c][tag];
        }
    }
}

// =====================================================================
// Kernel 6: emit the predicted path, chunks in parallel.
// grid (CHUNKS, BATCH), 1 wave.
// =====================================================================
__global__ __launch_bounds__(32) void emit_path(
    const unsigned char* __restrict__ bpW,
    const unsigned char* __restrict__ tagEndW,
    float* __restrict__ pred)              // [BATCH][SLEN] as float
{
    const int c = blockIdx.x, b = blockIdx.y;
    const int t = threadIdx.x;
    __shared__ unsigned char bpl[CLEN][TP];
    for (int idx = t; idx < CLEN * TP; idx += 32)
        bpl[idx >> 4][idx & 15] =
            bpW[((size_t)b * SLEN + CLEN * c + (idx >> 4)) * TP + (idx & 15)];
    __syncthreads();
    if (t == 0) {
        int tag = tagEndW[(size_t)b * CHUNKS + c];
        const int sFirst = (c == 0) ? 1 : CLEN * c;
        for (int s = CLEN * c + CLEN - 1; s >= sFirst; --s) {
            pred[(size_t)b * SLEN + s] = (float)tag;
            tag = bpl[s - CLEN * c][tag];
        }
        if (c == 0) pred[(size_t)b * SLEN + 0] = (float)tag;
    }
}

// =====================================================================
// Kernel 7: loss = mean(-ll); lengths passthrough as floats.
// =====================================================================
__global__ __launch_bounds__(64) void finalize_kernel(
    const float* __restrict__ ll, const int* __restrict__ lengths,
    float* __restrict__ out)
{
    __shared__ float sh[BATCH];
    const int i = threadIdx.x;
    sh[i] = ll[i];
    __syncthreads();
    if (i == 0) {
        float s = 0.0f;
        for (int j = 0; j < BATCH; ++j) s += sh[j];
        out[0] = -s / (float)BATCH;
    }
    out[1 + ROWS + i] = (float)lengths[i];
}

// =====================================================================
// Launch
// =====================================================================
extern "C" void kernel_launch(void* const* d_in, const int* in_sizes, int n_in,
                              void* d_out, int out_size, void* d_ws, size_t ws_size,
                              hipStream_t stream) {
    const float* encoder_out = (const float*)d_in[0]; // [B,S,H]
    const int*   labels      = (const int*)  d_in[1]; // [B,S]
    const int*   mask        = (const int*)  d_in[2]; // [B,S]
    const int*   lengths     = (const int*)  d_in[3]; // [B]
    const float* fc_w        = (const float*)d_in[4]; // [T,H]
    const float* fc_b        = (const float*)d_in[5]; // [T]
    const float* start_trans = (const float*)d_in[6]; // [T]
    const float* end_trans   = (const float*)d_in[7]; // [T]
    const float* trans       = (const float*)d_in[8]; // [T,T]

    float* out = (float*)d_out; // [0]=loss, [1..1+B*S)=pred, then lengths

    // ---------------- workspace layout ----------------
    char* ws = (char*)d_ws;
    size_t off = 0;
    float* em     = (float*)(ws + off); off += (size_t)ROWS * TP * sizeof(float);
    float* MW     = (float*)(ws + off); off += (size_t)BATCH * CHUNKS * TNUM * TP * sizeof(float);
    float* VW     = (float*)(ws + off); off += (size_t)BATCH * CHUNKS * TNUM * TP * sizeof(float);
    float* ventry = (float*)(ws + off); off += (size_t)BATCH * CHUNKS * TP * sizeof(float);
    float* ll     = (float*)(ws + off); off += (size_t)BATCH * sizeof(float);
    int*   lastTagW = (int*)(ws + off); off += (size_t)BATCH * sizeof(int);
    unsigned char* bpW     = (unsigned char*)(ws + off); off += (size_t)ROWS * TP;
    unsigned char* HcW     = (unsigned char*)(ws + off); off += (size_t)BATCH * CHUNKS * TP;
    unsigned char* tagEndW = (unsigned char*)(ws + off); off += (size_t)BATCH * CHUNKS;

    // 1) emissions GEMM (WMMA): 2048 tiles of 16 rows, 4 waves / block
    emissions_wmma<<<ROWS / 16 / 4, 128, 0, stream>>>(encoder_out, fc_w, fc_b, em);

    // 2) chunk transfer matrices (both semirings)
    chunk_mats<<<dim3(CHUNKS, BATCH), 160, 0, stream>>>(em, mask, trans, MW, VW);

    // 3) boundary sweep + denominator + numerator + viterbi entry vectors
    boundary_scan<<<BATCH, 64, 0, stream>>>(em, labels, mask, start_trans,
                                            end_trans, trans, MW, VW,
                                            ventry, ll, lastTagW);

    // 4) per-chunk backpointers + chunk backtrace maps
    bp_gen<<<dim3(CHUNKS, BATCH), 32, 0, stream>>>(em, mask, trans, ventry,
                                                   bpW, HcW);

    // 5) chain chunk maps -> boundary tags
    chain_kernel<<<BATCH, 32, 0, stream>>>(HcW, lastTagW, tagEndW);

    // 6) emit predictions, chunks in parallel
    emit_path<<<dim3(CHUNKS, BATCH), 32, 0, stream>>>(bpW, tagEndW, out + 1);

    // 7) loss reduction + lengths
    finalize_kernel<<<1, BATCH, 0, stream>>>(ll, lengths, out);
}
